// NeuralODE_10703058501936
// MI455X (gfx1250) — compile-verified, matrix operand order checked by reading, not run on hardware
//
#include <hip/hip_runtime.h>

typedef __attribute__((ext_vector_type(16))) __bf16 v16bf;
typedef __attribute__((ext_vector_type(8)))  __bf16 v8bf;
typedef __attribute__((ext_vector_type(8)))  float  v8f;

#define PITCH 264   // bf16 elements per LDS row (528B, 16B aligned)

// Dopri5 tableau
static __device__ const float Acoef[6][5] = {
    {0.f, 0.f, 0.f, 0.f, 0.f},
    {0.2f, 0.f, 0.f, 0.f, 0.f},
    {3.0f/40.0f, 9.0f/40.0f, 0.f, 0.f, 0.f},
    {44.0f/45.0f, -56.0f/15.0f, 32.0f/9.0f, 0.f, 0.f},
    {19372.0f/6561.0f, -25360.0f/2187.0f, 64448.0f/6561.0f, -212.0f/729.0f, 0.f},
    {9017.0f/3168.0f, -355.0f/33.0f, 46732.0f/5247.0f, 49.0f/176.0f, -5103.0f/18656.0f}};
static __device__ const float Ccoef[6] = {0.0f, 0.2f, 0.3f, 0.8f, 8.0f/9.0f, 1.0f};
static __device__ const float Bcoef[6] = {35.0f/384.0f, 0.0f, 500.0f/1113.0f,
                                          125.0f/192.0f, -2187.0f/6784.0f, 11.0f/84.0f};

// Load a 16x32 bf16 WMMA fragment (A row-major, or B via column-major storage)
// from an LDS tile with row pitch PITCH. ISA 7.12.2 layout: lane = row (M/N),
// K split into [koff, koff+8) and [koff+16, koff+24), koff = (lane>=16)*8.
__device__ inline v16bf ld_frag(const __bf16* base, int row16, int lane, int kbase) {
    int r    = row16 + (lane & 15);
    int koff = (lane >> 4) * 8;
    const __bf16* p = base + r * PITCH + kbase + koff;
    v8bf lo = *(const v8bf*)p;          // K = kbase+koff .. +7
    v8bf hi = *(const v8bf*)(p + 16);   // K = kbase+koff+16 .. +23
    v16bf f;
#pragma unroll
    for (int i = 0; i < 8; ++i) { f[i] = lo[i]; f[8 + i] = hi[i]; }
    return f;
}

__global__ __launch_bounds__(256)
void neural_ode_dopri5_wmma_kernel(const float* __restrict__ gx,
                                   const float* __restrict__ gW1,
                                   const float* __restrict__ gb1,
                                   const float* __restrict__ gb2,
                                   const float* __restrict__ gW2,
                                   const float* __restrict__ gW3,
                                   const float* __restrict__ gb3,
                                   float* __restrict__ gout) {
    extern __shared__ char smem[];
    float* sW1 = (float*)smem;            // 33*256
    float* sB1 = sW1 + 33 * 256;          // 256
    float* sB2 = sB1 + 256;               // 256
    float* sB3 = sB2 + 256;               // 32
    float* sZ2 = sB3 + 32;                // 256  raw z2 row from WMMA
    float* sG2 = sZ2 + 256;               // 256
    float* sX  = sG2 + 256;               // 32
    float* sXS = sX + 32;                 // 32
    float* sDX = sXS + 32;                // 32
    float* sKX = sDX + 32;                // 6*32
    float* sKS = sKX + 6 * 32;            // 6*4
    float* sSc = sKS + 24;                // [0]=tr [1]=frob [2]=logdet [3]=vfnA [4]=jfA [5]=stage vfn
    __bf16* sW2T = (__bf16*)(sSc + 8);    // 256 x PITCH  (W2 transposed: [n][k])
    __bf16* sW3T = sW2T + 256 * PITCH;    // 32  x PITCH  (W3 transposed: [n][k])
    __bf16* sT   = sW3T + 32 * PITCH;     // 48  x PITCH  rows 0-31: T1, row 32: h1, 33-47: 0
    __bf16* sT2  = sT + 48 * PITCH;       // 48  x PITCH  rows 0-31: T2, row 32: h2, 33-47: 0

    const int tid  = threadIdx.x;
    const int lane = tid & 31;
    const int wave = tid >> 5;

    // ---- stage in weights (W2/W3 transposed + bf16 for WMMA B operands) ----
    for (int e = tid; e < 33 * 256; e += 256) sW1[e] = gW1[e];
    sB1[tid] = gb1[tid];
    sB2[tid] = gb2[tid];
    if (tid < 32) sB3[tid] = gb3[tid];
    for (int e = tid; e < 256 * 256; e += 256) {
        int k = e >> 8, n = e & 255;
        sW2T[n * PITCH + k] = (__bf16)gW2[e];
    }
    for (int e = tid; e < 256 * 32; e += 256) {
        int k = e >> 5, n = e & 31;
        sW3T[n * PITCH + k] = (__bf16)gW3[e];
    }
    for (int e = tid; e < 15 * PITCH; e += 256) {   // zero pad rows 33..47
        sT [33 * PITCH + e] = (__bf16)0.0f;
        sT2[33 * PITCH + e] = (__bf16)0.0f;
    }
    if (tid < 32) sX[tid] = gx[(size_t)blockIdx.x * 32 + tid];
    if (tid == 0) { sSc[2] = 0.f; sSc[3] = 0.f; sSc[4] = 0.f; }
    __syncthreads();

    const float dt = 1.0f / 8.0f;
    for (int step = 0; step < 8; ++step) {
        float t0 = dt * (float)step;
        for (int st = 0; st < 6; ++st) {
            // stage state x_i = x + dt * sum_j A[i][j] k_j
            if (tid < 32) {
                float v = sX[tid];
                for (int j = 0; j < st; ++j) v += dt * Acoef[st][j] * sKX[j * 32 + tid];
                sXS[tid] = v;
            }
            if (tid == 0) { sSc[0] = 0.f; sSc[1] = 0.f; sSc[5] = 0.f; }
            __syncthreads();

            float teff = 1.0f - (t0 + Ccoef[st] * dt);

            // ---- layer 1 (VALU, tiny): z1 = [x,t]@W1 + b1; build A tile ----
            float z1 = sB1[tid];
#pragma unroll 8
            for (int d = 0; d < 32; ++d) z1 = fmaf(sXS[d], sW1[d * 256 + tid], z1);
            z1 = fmaf(teff, sW1[32 * 256 + tid], z1);
            float h1v = tanhf(z1);
            float g1v = 1.0f - h1v * h1v;
            sT[32 * PITCH + tid] = (__bf16)h1v;              // row 32 = h1
#pragma unroll 8
            for (int r = 0; r < 32; ++r)                     // rows 0..31 = T1
                sT[r * PITCH + tid] = (__bf16)(sW1[r * 256 + tid] * g1v);
            __syncthreads();

            // ---- WMMA phase 1a: z2 = h1 @ W2 (mt=2 tile row), 2 N-tiles/wave ----
            {
                int nt0 = wave * 2, nt1 = wave * 2 + 1;
                v8f acc0 = {}, acc1 = {};
#pragma unroll
                for (int kk = 0; kk < 8; ++kk) {
                    v16bf a  = ld_frag(sT, 32, lane, kk * 32);       // rows 32..47
                    v16bf b0 = ld_frag(sW2T, nt0 * 16, lane, kk * 32);
                    v16bf b1 = ld_frag(sW2T, nt1 * 16, lane, kk * 32);
                    acc0 = __builtin_amdgcn_wmma_f32_16x16x32_bf16(
                        false, a, false, b0, (short)0, acc0, false, false);
                    acc1 = __builtin_amdgcn_wmma_f32_16x16x32_bf16(
                        false, a, false, b1, (short)0, acc1, false, false);
                }
                if (lane < 16) {                 // local row 0 = VGPR0, lanes 0-15
                    sZ2[nt0 * 16 + lane] = acc0[0];
                    sZ2[nt1 * 16 + lane] = acc1[0];
                }
            }
            __syncthreads();

            // ---- h2 / g2 ----
            {
                float h2v = tanhf(sZ2[tid] + sB2[tid]);
                sG2[tid] = 1.0f - h2v * h2v;
                sT2[32 * PITCH + tid] = (__bf16)h2v;         // row 32 of phase-2 A
            }
            __syncthreads();

            // ---- WMMA phase 1b: T2 = (T1 @ W2) * g2; A frag shared per wave ----
            {
                int mt  = wave >> 2;             // waves 0-3 -> mt0, 4-7 -> mt1
                int ntb = (wave & 3) * 4;
                v8f acc[4] = {{}, {}, {}, {}};
#pragma unroll
                for (int kk = 0; kk < 8; ++kk) {
                    v16bf a = ld_frag(sT, mt * 16, lane, kk * 32);
#pragma unroll
                    for (int r = 0; r < 4; ++r) {
                        v16bf b = ld_frag(sW2T, (ntb + r) * 16, lane, kk * 32);
                        acc[r] = __builtin_amdgcn_wmma_f32_16x16x32_bf16(
                            false, a, false, b, (short)0, acc[r], false, false);
                    }
                }
#pragma unroll
                for (int r = 0; r < 4; ++r) {
                    int n = (ntb + r) * 16 + (lane & 15);
                    float g = sG2[n];
#pragma unroll
                    for (int v = 0; v < 8; ++v) {
                        int m = mt * 16 + (lane >> 4) * 8 + v;
                        sT2[m * PITCH + n] = (__bf16)(acc[r][v] * g);
                    }
                }
            }
            __syncthreads();

            // ---- WMMA phase 2: [T2; h2] @ W3 -> J (mt 0,1) and dxdt (mt 2) ----
            if (wave < 6) {
                int mt = wave >> 1, nt = wave & 1;
                v8f acc = {};
#pragma unroll
                for (int kk = 0; kk < 8; ++kk) {
                    v16bf a = ld_frag(sT2,  mt * 16, lane, kk * 32);
                    v16bf b = ld_frag(sW3T, nt * 16, lane, kk * 32);
                    acc = __builtin_amdgcn_wmma_f32_16x16x32_bf16(
                        false, a, false, b, (short)0, acc, false, false);
                }
                int n = nt * 16 + (lane & 15);
                if (mt < 2) {                    // J tile: trace + Frobenius
                    float fr = 0.f, trp = 0.f;
#pragma unroll
                    for (int v = 0; v < 8; ++v) {
                        int m = mt * 16 + (lane >> 4) * 8 + v;
                        fr += acc[v] * acc[v];
                        if (m == n) trp += acc[v];
                    }
                    atomicAdd(&sSc[1], fr);
                    if (mt == nt) atomicAdd(&sSc[0], trp);
                } else if (lane < 16) {          // dxdt row (local row 0)
                    float dx = -(acc[0] + sB3[n]);
                    sDX[n] = dx;
                    atomicAdd(&sSc[5], dx * dx);
                }
            }
            __syncthreads();

            // record stage derivatives
            if (tid < 32) sKX[st * 32 + tid] = sDX[tid];
            if (tid == 0) {
                sKS[st * 4 + 0] = -sSc[0];   // k_logdet = -trace(J)
                sKS[st * 4 + 1] = sSc[5];    // k_vfnorm = |dxdt|^2
                sKS[st * 4 + 2] = sSc[1];    // k_frob   = sum J^2
            }
            __syncthreads();
        }

        // combine stages: y += dt * sum_i B[i] k_i
        if (tid < 32) {
            float v = sX[tid];
            for (int j = 0; j < 6; ++j) v += dt * Bcoef[j] * sKX[j * 32 + tid];
            sX[tid] = v;
        }
        if (tid == 0) {
            float a0 = 0.f, a1 = 0.f, a2 = 0.f;
            for (int j = 0; j < 6; ++j) {
                a0 += Bcoef[j] * sKS[j * 4 + 0];
                a1 += Bcoef[j] * sKS[j * 4 + 1];
                a2 += Bcoef[j] * sKS[j * 4 + 2];
            }
            sSc[2] += dt * a0; sSc[3] += dt * a1; sSc[4] += dt * a2;
        }
        __syncthreads();
    }

    // output: (B, 35) = [z(32), log_det, vf_norm, jac_frob]
    if (tid < 32) gout[(size_t)blockIdx.x * 35 + tid] = sX[tid];
    if (tid >= 32 && tid < 35) gout[(size_t)blockIdx.x * 35 + tid] = sSc[tid - 32 + 2];
}

static constexpr size_t SMEM_BYTES =
    (33 * 256 + 256 + 256 + 32 + 256 + 256 + 32 + 32 + 32 + 6 * 32 + 24 + 8) * sizeof(float)
    + (size_t)((256 + 32 + 48 + 48) * PITCH) * sizeof(__bf16);   // 242048 B < 320 KB/WGP

extern "C" void kernel_launch(void* const* d_in, const int* in_sizes, int n_in,
                              void* d_out, int out_size, void* d_ws, size_t ws_size,
                              hipStream_t stream) {
    (void)n_in; (void)out_size; (void)d_ws; (void)ws_size;
    const float* x  = (const float*)d_in[0];
    const float* W1 = (const float*)d_in[1];
    const float* b1 = (const float*)d_in[2];
    const float* W2 = (const float*)d_in[3];
    const float* b2 = (const float*)d_in[4];
    const float* W3 = (const float*)d_in[5];
    const float* b3 = (const float*)d_in[6];
    float* out = (float*)d_out;

    int nB = in_sizes[0] / 32;  // 512 samples, one workgroup each

    // CDNA5 allows 320KB LDS per workgroup; opt in for >64KB dynamic LDS.
    hipStreamCaptureStatus cap = hipStreamCaptureStatusNone;
    (void)hipStreamIsCapturing(stream, &cap);
    if (cap == hipStreamCaptureStatusNone) {
        (void)hipFuncSetAttribute(
            reinterpret_cast<const void*>(neural_ode_dopri5_wmma_kernel),
            hipFuncAttributeMaxDynamicSharedMemorySize, (int)SMEM_BYTES);
    }

    neural_ode_dopri5_wmma_kernel<<<nB, 256, SMEM_BYTES, stream>>>(
        x, W1, b1, b2, W2, W3, b3, out);
}
